// Head_74414603370664
// MI455X (gfx1250) — matrix operand order repaired and checked by reading, not compiled, not run on hardware
//
#include <hip/hip_runtime.h>

typedef __attribute__((ext_vector_type(16))) __bf16 v16bf;
typedef __attribute__((ext_vector_type(8)))  __bf16 bfx8;
typedef __attribute__((ext_vector_type(8)))  float  v8f;

#define B_DIM 8
#define T_DIM 2048
#define D_DIM 1024
#define H_DIM 128

static __device__ __forceinline__ v16bf cat16(bfx8 lo, bfx8 hi) {
    v16bf r;
    #pragma unroll
    for (int i = 0; i < 8; ++i) { r[i] = lo[i]; r[i + 8] = hi[i]; }
    return r;
}

static __device__ __forceinline__ v8f zero8() {
    v8f z = {0.f, 0.f, 0.f, 0.f, 0.f, 0.f, 0.f, 0.f};
    return z;
}

// ---------------------------------------------------------------------------
// Kernel 0: weight pre-pack.  W[D,H] f32 -> bf16 in WMMA B-fragment (contiguous-k order used here)
// layout wpack[mat][kc/32][colTile][lane][16], element j of a lane's slice is
// K = kc + j + 16*(lane>=16), N = colTile*16 + (lane&15).  After this, every
// B-fragment in the projection is two contiguous 16B loads.
// grid = (D/32, H/16, 3), block = 32.
// ---------------------------------------------------------------------------
__global__ __launch_bounds__(32) void wpack_kernel(
    const float* __restrict__ Wq,
    const float* __restrict__ Wk,
    const float* __restrict__ Wv,
    __bf16* __restrict__ wpack)
{
    const int lane = threadIdx.x & 31;
    const int half = lane >> 4;
    const int l16  = lane & 15;
    const int chunk = blockIdx.x;   // kc / 32
    const int ct    = blockIdx.y;   // col tile 0..7
    const int mat   = blockIdx.z;   // 0=Q 1=K 2=V
    const float* __restrict__ W = (mat == 0) ? Wq : ((mat == 1) ? Wk : Wv);

    __bf16* dst = wpack +
        (((size_t)mat * (D_DIM / 32) + chunk) * (H_DIM / 16) + ct) * 512 + lane * 16;
    const int kcbase = chunk * 32 + 16 * half;
    const int col    = ct * 16 + l16;
    #pragma unroll
    for (int j = 0; j < 16; ++j)
        dst[j] = (__bf16)W[(size_t)(kcbase + j) * H_DIM + col];
}

// ---------------------------------------------------------------------------
// Kernel 1: fused QKV projection.  One block per 16-row tile of x; the tile is
// staged once into LDS as bf16 (32KB of the 320KB WGP pool), then each of the
// 8 waves computes its 16-col tile of Q, K and V (3 accumulators) -> x is
// read from global exactly once.  V is stored transposed [b][h][t].
// grid = (B*T/16), block = 256.
// ---------------------------------------------------------------------------
__global__ __launch_bounds__(256) void qkv_proj_kernel(
    const float* __restrict__ x,
    const __bf16* __restrict__ wpack,
    __bf16* __restrict__ qout,
    __bf16* __restrict__ kout,
    __bf16* __restrict__ vtout)
{
    __shared__ __align__(16) __bf16 xlds[16][D_DIM];   // 32 KB

    const int tid  = threadIdx.x;
    const int lane = tid & 31;
    const int wv   = tid >> 5;     // col tile 0..7
    const int half = lane >> 4;
    const int l16  = lane & 15;
    const int rowbase = blockIdx.x * 16;
    const int colbase = wv * 16;

    // Cooperative stage: 16 rows x 1024 f32 -> bf16 LDS (8 floats / thread / iter).
    for (int i = tid; i < 16 * (D_DIM / 8); i += 256) {
        const int row = i >> 7;             // 128 groups of 8 per row
        const int col = (i & 127) << 3;
        const float* xp = x + (size_t)(rowbase + row) * D_DIM + col;
        float4 f0 = *(const float4*)xp;
        float4 f1 = *(const float4*)(xp + 4);
        bfx8 bv;
        bv[0] = (__bf16)f0.x; bv[1] = (__bf16)f0.y; bv[2] = (__bf16)f0.z; bv[3] = (__bf16)f0.w;
        bv[4] = (__bf16)f1.x; bv[5] = (__bf16)f1.y; bv[6] = (__bf16)f1.z; bv[7] = (__bf16)f1.w;
        *(bfx8*)&xlds[row][col] = bv;
    }
    __syncthreads();

    const size_t matstride = (size_t)D_DIM * H_DIM;    // elements per packed matrix
    v8f accq = zero8(), acck = zero8(), accv = zero8();

    for (int kc = 0; kc < D_DIM; kc += 32) {
        // A fragment from LDS: lane = row M, element j -> K = kc + j + 16*half
        const __bf16* ap = &xlds[l16][kc + 16 * half];
        v16bf a = cat16(*(const bfx8*)ap, *(const bfx8*)(ap + 8));

        // Packed B fragments: contiguous 32B per lane per matrix.
        const size_t base = (((size_t)(kc >> 5)) * (H_DIM / 16) + wv) * 512 + lane * 16;
        const __bf16* pq = wpack + base;
        const __bf16* pk = wpack + matstride + base;
        const __bf16* pv = wpack + 2 * matstride + base;
        v16bf bq = cat16(*(const bfx8*)pq, *(const bfx8*)(pq + 8));
        v16bf bk = cat16(*(const bfx8*)pk, *(const bfx8*)(pk + 8));
        v16bf bvv = cat16(*(const bfx8*)pv, *(const bfx8*)(pv + 8));

        accq = __builtin_amdgcn_wmma_f32_16x16x32_bf16(
            false, a, false, bq, (short)0, accq, false, false);
        acck = __builtin_amdgcn_wmma_f32_16x16x32_bf16(
            false, a, false, bk, (short)0, acck, false, false);
        accv = __builtin_amdgcn_wmma_f32_16x16x32_bf16(
            false, a, false, bvv, (short)0, accv, false, false);
    }

    // C layout: element r -> M = r + 8*half, N = l16.
    #pragma unroll
    for (int r = 0; r < 8; ++r) {
        const int m = r + 8 * half;
        qout[(size_t)(rowbase + m) * H_DIM + colbase + l16] = (__bf16)accq[r];
        kout[(size_t)(rowbase + m) * H_DIM + colbase + l16] = (__bf16)acck[r];
    }
    const int bidx = rowbase / T_DIM;
    const int t0   = rowbase % T_DIM;
    #pragma unroll
    for (int r = 0; r < 8; ++r) {
        const int m = r + 8 * half;
        vtout[((size_t)bidx * H_DIM + colbase + l16) * T_DIM + (t0 + m)] = (__bf16)accv[r];
    }
}

// ---------------------------------------------------------------------------
// Kernel 2: causal flash attention.  One wave owns 16 queries x H=128.
// Iterates keys in 32-chunks with online softmax; P converted C-layout ->
// A-layout through a private 1KB LDS slice (per-wave s_wait_dscnt sync, no
// block barrier — trip counts diverge across waves).
// grid = (T/64, B), block = 128 (4 waves).
// ---------------------------------------------------------------------------
__global__ __launch_bounds__(128) void flash_attn_kernel(
    const __bf16* __restrict__ q,
    const __bf16* __restrict__ k,
    const __bf16* __restrict__ vt,
    float* __restrict__ out)
{
    __shared__ __align__(16) __bf16 plds[4][16][32];

    const int lane = threadIdx.x & 31;
    const int wv   = threadIdx.x >> 5;
    const int half = lane >> 4;
    const int l16  = lane & 15;
    const int b    = blockIdx.y;
    const int qbase = (blockIdx.x * 4 + wv) * 16;

    const __bf16* __restrict__ qb  = q  + (size_t)b * T_DIM * H_DIM;
    const __bf16* __restrict__ kbp = k  + (size_t)b * T_DIM * H_DIM;
    const __bf16* __restrict__ vtb = vt + (size_t)b * H_DIM * T_DIM;
    float* __restrict__ ob = out + (size_t)b * T_DIM * H_DIM;

    // Q A-fragments for the 4 h-chunks, kept in registers for the whole loop.
    v16bf qf[4];
    #pragma unroll
    for (int hc = 0; hc < 4; ++hc) {
        const __bf16* p0 = qb + (size_t)(qbase + l16) * H_DIM + hc * 32 + 16 * half;
        qf[hc] = cat16(*(const bfx8*)p0, *(const bfx8*)(p0 + 8));
    }

    const float scale = 0.08838834764831845f;  // 1/sqrt(128)
    float mrow[8], lrow[8], corr[8];
    v8f o[8];
    #pragma unroll
    for (int r = 0; r < 8; ++r) { mrow[r] = -__builtin_inff(); lrow[r] = 0.f; }
    #pragma unroll
    for (int ht = 0; ht < 8; ++ht) o[ht] = zero8();

    for (int kb0 = 0; kb0 < qbase + 16; kb0 += 32) {
        const bool have1 = (kb0 + 16 <= qbase);   // wave-uniform

        if (kb0 + 32 < T_DIM)
            __builtin_prefetch(kbp + (size_t)(kb0 + 32 + l16) * H_DIM, 0, 1);

        // S = Q K^T for two 16-key subtiles (contract H=128 in 4 chunks).
        v8f s0 = zero8(), s1 = zero8();
        #pragma unroll
        for (int hc = 0; hc < 4; ++hc) {
            const __bf16* p = kbp + (size_t)(kb0 + l16) * H_DIM + hc * 32 + 16 * half;
            v16bf kf = cat16(*(const bfx8*)p, *(const bfx8*)(p + 8));
            s0 = __builtin_amdgcn_wmma_f32_16x16x32_bf16(
                false, qf[hc], false, kf, (short)0, s0, false, false);
        }
        if (have1) {
            #pragma unroll
            for (int hc = 0; hc < 4; ++hc) {
                const __bf16* p = kbp + (size_t)(kb0 + 16 + l16) * H_DIM + hc * 32 + 16 * half;
                v16bf kf = cat16(*(const bfx8*)p, *(const bfx8*)(p + 8));
                s1 = __builtin_amdgcn_wmma_f32_16x16x32_bf16(
                    false, qf[hc], false, kf, (short)0, s1, false, false);
            }
        }

        // Causal mask + online softmax (row m = r + 8*half, key col = l16).
        #pragma unroll
        for (int r = 0; r < 8; ++r) {
            const int qpos = qbase + r + 8 * half;
            float v0 = s0[r] * scale;
            v0 = (kb0 + l16 > qpos) ? -__builtin_inff() : v0;
            float v1 = -__builtin_inff();
            if (have1) {
                v1 = s1[r] * scale;
                v1 = (kb0 + 16 + l16 > qpos) ? -__builtin_inff() : v1;
            }
            float mx = fmaxf(v0, v1);
            mx = fmaxf(mx, __shfl_xor(mx, 1, 32));
            mx = fmaxf(mx, __shfl_xor(mx, 2, 32));
            mx = fmaxf(mx, __shfl_xor(mx, 4, 32));
            mx = fmaxf(mx, __shfl_xor(mx, 8, 32));
            const float newm = fmaxf(mrow[r], mx);
            const float c = __expf(mrow[r] - newm);
            mrow[r] = newm;
            corr[r] = c;
            const float e0 = __expf(v0 - newm);
            const float e1 = have1 ? __expf(v1 - newm) : 0.f;
            float rs = e0 + e1;
            rs += __shfl_xor(rs, 1, 32);
            rs += __shfl_xor(rs, 2, 32);
            rs += __shfl_xor(rs, 4, 32);
            rs += __shfl_xor(rs, 8, 32);
            lrow[r] = lrow[r] * c + rs;
            const int m = r + 8 * half;
            plds[wv][m][l16]      = (__bf16)e0;
            plds[wv][m][16 + l16] = (__bf16)e1;
        }

        // Rescale accumulators before adding this chunk's P*V.
        #pragma unroll
        for (int ht = 0; ht < 8; ++ht) {
            #pragma unroll
            for (int r = 0; r < 8; ++r)
                o[ht][r] *= corr[r];
        }

        // Wave-private LDS round trip: wait for DS writes to land.
        asm volatile("s_wait_dscnt 0" ::: "memory");

        // P as A-fragment (16 rows x 32 keys).
        const __bf16* pl = &plds[wv][l16][0];
        v16bf pf = cat16(*(const bfx8*)(pl + 16 * half),
                         *(const bfx8*)(pl + 16 * half + 8));

        // O += P * V over the 8 h-column tiles (V transposed: contiguous in t).
        #pragma unroll
        for (int ht = 0; ht < 8; ++ht) {
            const __bf16* vp = vtb + (size_t)(ht * 16 + l16) * T_DIM + kb0 + 16 * half;
            v16bf vf = cat16(*(const bfx8*)vp, *(const bfx8*)(vp + 8));
            o[ht] = __builtin_amdgcn_wmma_f32_16x16x32_bf16(
                false, pf, false, vf, (short)0, o[ht], false, false);
        }
    }

    #pragma unroll
    for (int r = 0; r < 8; ++r) {
        const int m = r + 8 * half;
        const float inv = 1.0f / lrow[r];
        #pragma unroll
        for (int ht = 0; ht < 8; ++ht)
            ob[(size_t)(qbase + m) * H_DIM + ht * 16 + l16] = o[ht][r] * inv;
    }
}

extern "C" void kernel_launch(void* const* d_in, const int* in_sizes, int n_in,
                              void* d_out, int out_size, void* d_ws, size_t ws_size,
                              hipStream_t stream) {
    const float* x  = (const float*)d_in[0];
    const float* Wq = (const float*)d_in[1];
    const float* Wk = (const float*)d_in[2];
    const float* Wv = (const float*)d_in[3];
    float* out = (float*)d_out;

    // Workspace: q, k (row-major bf16), v transposed (bf16), packed weights.
    // 3 * 4MB + 768KB = ~12.75MB.
    const size_t nqkv = (size_t)B_DIM * T_DIM * H_DIM;
    __bf16* qb    = (__bf16*)d_ws;
    __bf16* kb    = qb + nqkv;
    __bf16* vt    = kb + nqkv;
    __bf16* wpack = vt + nqkv;

    dim3 gW(D_DIM / 32, H_DIM / 16, 3);
    wpack_kernel<<<gW, 32, 0, stream>>>(Wq, Wk, Wv, wpack);

    dim3 gA((B_DIM * T_DIM) / 16, 1, 1);
    qkv_proj_kernel<<<gA, 256, 0, stream>>>(x, wpack, qb, kb, vt);

    dim3 gB(T_DIM / 64, B_DIM, 1);
    flash_attn_kernel<<<gB, 128, 0, stream>>>(qb, kb, vt, out);
}
